// Attention_50938312131153
// MI455X (gfx1250) — matrix operand order repaired and checked by reading, not compiled, not run on hardware
//
#include <hip/hip_runtime.h>
#include <hip/hip_bf16.h>

// ---------------------------------------------------------------------------
// Fused attention: Q/K/V projection + softmax(QK^T/sqrt(E)) V  on MI455X
// bf16 WMMA (v_wmma_f32_16x16x32_bf16) everywhere, f32 accumulate.
// Q tile staged to LDS via GLOBAL_LOAD_ASYNC_TO_LDS_B128 when available.
// ---------------------------------------------------------------------------

typedef __attribute__((ext_vector_type(16))) __bf16 bf16x16;
typedef __attribute__((ext_vector_type(8)))  __bf16 bf16x8;
typedef __attribute__((ext_vector_type(8)))  float  f32x8;
typedef __attribute__((ext_vector_type(4)))  int    i32x4;

#define DIM        1024      // INPUT_DIM == EMBED_DIM
#define SEQ        2048
#define BATCH      4
#define ROWS_TOTAL 8192      // BATCH * SEQ

// ---- workspace layout (bytes) ----
#define XB_BYTES   (3ull * ROWS_TOTAL * DIM * 2)            // bf16 q,k,v
#define WT_BYTES   (3ull * DIM * DIM * 2)                   // bf16 W^T
#define QB_BYTES   (1ull * ROWS_TOTAL * DIM * 2)
#define XB_OFF     0ull
#define WT_OFF     (XB_OFF + XB_BYTES)
#define QB_OFF     (WT_OFF + WT_BYTES)
#define KB_OFF     (QB_OFF + QB_BYTES)
#define VT_OFF     (KB_OFF + QB_BYTES)

// ---- optional CDNA5 async global->LDS copy path ----
#if defined(__has_builtin)
#if __has_builtin(__builtin_amdgcn_global_load_async_to_lds_b128) && \
    __has_builtin(__builtin_amdgcn_s_wait_asynccnt)
#define USE_ASYNC_COPY 1
#endif
#endif

#ifdef USE_ASYNC_COPY
// Builtin signature (from compiler diagnostic): param0 = as(1) int4*,
// so pass typed vector pointers: global src in as(1), LDS dest in as(3).
typedef __attribute__((address_space(1))) i32x4 as1_i32x4;
typedef __attribute__((address_space(3))) i32x4 as3_i32x4;
__device__ __forceinline__ void async_copy_b128(const void* g, void* l) {
    // as(1) = full 64-bit global address; as(3) = low 32 bits of the generic
    // LDS address (LDS aperture keeps the byte offset in addr[31:0]).
    __builtin_amdgcn_global_load_async_to_lds_b128(
        (as1_i32x4*)(unsigned long long)g,
        (as3_i32x4*)(unsigned int)(unsigned long long)l,
        /*imm offset=*/0, /*cpol=*/0);
}
#endif

// ---------------------------------------------------------------------------
// WMMA fragment helpers (CDNA5 wave32 layouts, 16x16x32 bf16)
// ---------------------------------------------------------------------------

// A matrix 16x32 (MxK), row-major source with given stride (elements).
// lanes 0-15: K = {0..7, 16..23};  lanes 16-31: K = {8..15, 24..31}
__device__ __forceinline__ bf16x16 load_frag_a(const __bf16* p, int stride, int lane) {
    const int m  = lane & 15;
    const int kh = (lane >> 4) << 3;              // 0 or 8
    const __bf16* q = p + (size_t)m * stride + kh;
    union { bf16x16 v; bf16x8 h[2]; } u;
    u.h[0] = *(const bf16x8*)(q);                 // K = kh .. kh+7
    u.h[1] = *(const bf16x8*)(q + 16);            // K = kh+16 .. kh+23
    return u.v;
}

// B matrix 32x16 (KxN) where B[k][n] = M2[n][k]  (transposed operand, NT GEMM).
// lane = N column; lanes 0-15: K = {0..7, 16..23};  lanes 16-31: K = {8..15, 24..31}
__device__ __forceinline__ bf16x16 load_frag_bt(const __bf16* p, int stride, int lane) {
    const int n  = lane & 15;
    const int kh = (lane >> 4) << 3;              // 0 or 8
    const __bf16* q = p + (size_t)n * stride + kh;
    union { bf16x16 v; bf16x8 h[2]; } u;
    u.h[0] = *(const bf16x8*)(q);
    u.h[1] = *(const bf16x8*)(q + 16);
    return u.v;
}

__device__ __forceinline__ f32x8 wmma_bf16(bf16x16 a, bf16x16 b, f32x8 c) {
    return __builtin_amdgcn_wmma_f32_16x16x32_bf16(
        /*neg_a=*/false, a, /*neg_b=*/false, b,
        /*c_mod=*/(short)0, c, /*reuse_a=*/false, /*reuse_b=*/false);
}

// ---------------------------------------------------------------------------
// Kernel 1: f32 -> bf16 conversion of q,k,v
// ---------------------------------------------------------------------------
__global__ __launch_bounds__(256) void convert_x_kernel(
    const float* __restrict__ q, const float* __restrict__ k,
    const float* __restrict__ v, __bf16* __restrict__ xb)
{
    const float* src = (blockIdx.z == 0) ? q : (blockIdx.z == 1) ? k : v;
    __bf16* dst = xb + (size_t)blockIdx.z * ((size_t)ROWS_TOTAL * DIM);
    size_t i = (size_t)blockIdx.x * blockDim.x + threadIdx.x;
    if (i < (size_t)ROWS_TOTAL * DIM) dst[i] = (__bf16)src[i];
}

// ---------------------------------------------------------------------------
// Kernel 2: W[d][e] f32 -> Wt[e][d] bf16 (LDS-tiled transpose)
// ---------------------------------------------------------------------------
__global__ __launch_bounds__(256) void transpose_w_kernel(
    const float* __restrict__ wq, const float* __restrict__ wk,
    const float* __restrict__ wv, __bf16* __restrict__ wt)
{
    const float* W = (blockIdx.z == 0) ? wq : (blockIdx.z == 1) ? wk : wv;
    __bf16* Wt = wt + (size_t)blockIdx.z * ((size_t)DIM * DIM);
    __shared__ float tile[32][33];
    const int tx = threadIdx.x & 31, ty = threadIdx.x >> 5;   // 32 x 8
    const int d0 = blockIdx.x * 32, e0 = blockIdx.y * 32;
#pragma unroll
    for (int r = ty; r < 32; r += 8)
        tile[r][tx] = W[(size_t)(d0 + r) * DIM + e0 + tx];
    __syncthreads();
#pragma unroll
    for (int r = ty; r < 32; r += 8)
        Wt[(size_t)(e0 + r) * DIM + d0 + tx] = (__bf16)tile[tx][r];
}

// ---------------------------------------------------------------------------
// Kernel 3: projection GEMM  out[s][e] = sum_d x[s][d] * Wt[e][d]
//   z=0 -> Qb (scaled by 1/32 = 1/sqrt(1024)),  z=1 -> Kb,  z=2 -> Vt[e][s]
// block = 256 threads (8 waves); block tile 128x64; wave tile 16x64.
// ---------------------------------------------------------------------------
__global__ __launch_bounds__(256) void proj_gemm_kernel(
    const __bf16* __restrict__ xb, const __bf16* __restrict__ wt,
    __bf16* __restrict__ Qb, __bf16* __restrict__ Kb, __bf16* __restrict__ Vt)
{
    const int z    = blockIdx.z;
    const __bf16* X = xb + (size_t)z * ((size_t)ROWS_TOTAL * DIM);
    const __bf16* W = wt + (size_t)z * ((size_t)DIM * DIM);
    const int lane = threadIdx.x & 31;
    const int w    = threadIdx.x >> 5;
    const int row0 = blockIdx.x * 128 + w * 16;
    const int col0 = blockIdx.y * 64;

    f32x8 acc[4] = {};
    const __bf16* arow = X + (size_t)row0 * DIM;

    for (int k0 = 0; k0 < DIM; k0 += 32) {
        __builtin_prefetch(arow + k0 + 128, 0, 3);     // global_prefetch_b8 (near)
        bf16x16 a = load_frag_a(arow + k0, DIM, lane);
#pragma unroll
        for (int t = 0; t < 4; ++t) {
            bf16x16 bm = load_frag_bt(W + (size_t)(col0 + 16 * t) * DIM + k0, DIM, lane);
            acc[t] = wmma_bf16(a, bm, acc[t]);
        }
    }

    const float qscale = (z == 0) ? 0.03125f : 1.0f;   // fold 1/sqrt(E) into Q
    const int rsub = 8 * (lane >> 4);
    const int cidx = lane & 15;
#pragma unroll
    for (int t = 0; t < 4; ++t) {
#pragma unroll
        for (int i = 0; i < 8; ++i) {
            const int row = row0 + rsub + i;
            const int col = col0 + 16 * t + cidx;
            const __bf16 val = (__bf16)(acc[t][i] * qscale);
            if (z == 0)      Qb[(size_t)row * DIM + col] = val;
            else if (z == 1) Kb[(size_t)row * DIM + col] = val;
            else {           // V stored transposed per batch: Vt[b][e][s]
                const int bb = row >> 11, s = row & (SEQ - 1);
                Vt[((size_t)bb << 21) + (size_t)col * SEQ + s] = val;
            }
        }
    }
}

// ---------------------------------------------------------------------------
// Kernel 4: flash attention.
//   grid = (SEQ/32, BATCH); block = 256 threads (8 waves).
//   Query tile = 32 rows (staged in LDS); key blocks of 64; online softmax.
//   Scores phase: wave w owns scores subtile (16*(w&1), 16*(w>>1)).
//   PV phase:     wave w owns output rows 16*(w&1).., cols 256*(w>>1)..
// ---------------------------------------------------------------------------
__global__ __launch_bounds__(256) void flash_kernel(
    const __bf16* __restrict__ Qb, const __bf16* __restrict__ Kb,
    const __bf16* __restrict__ Vt, float* __restrict__ Out)
{
    const int bidx = blockIdx.y;
    const int q0   = blockIdx.x * 32;
    const int tid  = threadIdx.x;
    const int lane = tid & 31;
    const int w    = tid >> 5;

    __shared__ __align__(32) __bf16 sQ[32][DIM];     // 64 KB staged Q tile
    __shared__ __align__(32) float  sS[32][64];
    __shared__ __align__(32) __bf16 sP[32][64];
    __shared__ float sM[32], sL[32], sA[32];

    const int sr = 16 * (w & 1);      // scores subtile row
    const int sc = 16 * (w >> 1);     // scores subtile col
    const int rh = 16 * (w & 1);      // PV row half
    const int cq = 256 * (w >> 1);    // PV col quarter (within E)
    const int rsub = 8 * (lane >> 4);
    const int cidx = lane & 15;

    // ---- stage Q tile (32 x 1024 bf16) into LDS: 4096 x b128 chunks ----
#pragma unroll
    for (int j = 0; j < 16; ++j) {
        const int c  = tid + 256 * j;        // chunk id
        const int r  = c >> 7;               // 128 chunks per row
        const int cc = (c & 127) * 8;        // element offset in row
        const __bf16* g = Qb + (size_t)(bidx * SEQ + q0 + r) * DIM + cc;
#ifdef USE_ASYNC_COPY
        async_copy_b128(g, &sQ[r][cc]);
#else
        *(bf16x8*)&sQ[r][cc] = *(const bf16x8*)g;
#endif
    }
#ifdef USE_ASYNC_COPY
    __builtin_amdgcn_s_wait_asynccnt(0);
#endif
    if (tid < 32) { sM[tid] = -3.0e38f; sL[tid] = 0.0f; }
    __syncthreads();

    f32x8 acc[16] = {};
    const __bf16* vbase = Vt + ((size_t)bidx << 21);

    for (int kb = 0; kb < SEQ / 64; ++kb) {
        const int s0 = kb * 64;

        // ---- scores subtile: Q(16xE) * K^T(Ex16), E = 1024 ----
        f32x8 sacc = {};
        const __bf16* krow = Kb + (size_t)(bidx * SEQ + s0 + sc) * DIM;
#pragma unroll 4
        for (int e0 = 0; e0 < DIM; e0 += 32) {
            bf16x16 a  = load_frag_a(&sQ[sr][e0], DIM, lane);   // ds loads
            bf16x16 bm = load_frag_bt(krow + e0, DIM, lane);
            sacc = wmma_bf16(a, bm, sacc);
        }
#pragma unroll
        for (int i = 0; i < 8; ++i)
            sS[sr + rsub + i][sc + cidx] = sacc[i];
        __syncthreads();

        // ---- online softmax update, one thread per query row ----
        if (tid < 32) {
            float m = sM[tid], mn = m;
#pragma unroll
            for (int j = 0; j < 64; ++j) mn = fmaxf(mn, sS[tid][j]);
            const float alpha = __expf(m - mn);
            float l = sL[tid] * alpha;
#pragma unroll
            for (int j = 0; j < 64; ++j) {
                const float p = __expf(sS[tid][j] - mn);
                sP[tid][j] = (__bf16)p;
                l += p;
            }
            sM[tid] = mn; sL[tid] = l; sA[tid] = alpha;
        }
        __syncthreads();

        // ---- rescale accumulators by alpha, then O += P * V_block ----
        float av[8];
#pragma unroll
        for (int i = 0; i < 8; ++i) av[i] = sA[rh + rsub + i];
#pragma unroll
        for (int t = 0; t < 16; ++t)
#pragma unroll
            for (int i = 0; i < 8; ++i) acc[t][i] *= av[i];

        bf16x16 pa[2];
#pragma unroll
        for (int kk = 0; kk < 2; ++kk)
            pa[kk] = load_frag_a(&sP[rh][kk * 32], 64, lane);   // ds loads
#pragma unroll
        for (int t = 0; t < 16; ++t) {
#pragma unroll
            for (int kk = 0; kk < 2; ++kk) {
                bf16x16 bv = load_frag_bt(
                    vbase + (size_t)(cq + 16 * t) * SEQ + (s0 + kk * 32), SEQ, lane);
                acc[t] = wmma_bf16(pa[kk], bv, acc[t]);
            }
        }
        __syncthreads();   // protect sS/sP for next iteration
    }

    // ---- finalize: divide by l, store f32 ----
    float linv[8];
#pragma unroll
    for (int i = 0; i < 8; ++i) linv[i] = 1.0f / sL[rh + rsub + i];
    float* orow = Out + (size_t)(bidx * SEQ + q0 + rh) * DIM;
#pragma unroll
    for (int t = 0; t < 16; ++t)
#pragma unroll
        for (int i = 0; i < 8; ++i)
            orow[(size_t)(rsub + i) * DIM + cq + 16 * t + cidx] = acc[t][i] * linv[i];
}

// ---------------------------------------------------------------------------
extern "C" void kernel_launch(void* const* d_in, const int* in_sizes, int n_in,
                              void* d_out, int out_size, void* d_ws, size_t ws_size,
                              hipStream_t stream) {
    const float* q  = (const float*)d_in[0];
    const float* k  = (const float*)d_in[1];
    const float* v  = (const float*)d_in[2];
    const float* wq = (const float*)d_in[3];
    const float* wk = (const float*)d_in[4];
    const float* wv = (const float*)d_in[5];
    float* out = (float*)d_out;

    char* ws = (char*)d_ws;
    __bf16* xb = (__bf16*)(ws + XB_OFF);
    __bf16* wt = (__bf16*)(ws + WT_OFF);
    __bf16* Qb = (__bf16*)(ws + QB_OFF);
    __bf16* Kb = (__bf16*)(ws + KB_OFF);
    __bf16* Vt = (__bf16*)(ws + VT_OFF);

    // 1) convert q,k,v to bf16
    {
        dim3 grid((ROWS_TOTAL * DIM) / 256, 1, 3);
        convert_x_kernel<<<grid, 256, 0, stream>>>(q, k, v, xb);
    }
    // 2) transpose weights to bf16 W^T
    {
        dim3 grid(DIM / 32, DIM / 32, 3);
        transpose_w_kernel<<<grid, 256, 0, stream>>>(wq, wk, wv, wt);
    }
    // 3) projections (Q scaled by 1/sqrt(E); V stored transposed)
    {
        dim3 grid(ROWS_TOTAL / 128, DIM / 64, 3);
        proj_gemm_kernel<<<grid, 256, 0, stream>>>(xb, wt, Qb, Kb, Vt);
    }
    // 4) flash attention
    {
        dim3 grid(SEQ / 32, BATCH);
        flash_kernel<<<grid, 256, 0, stream>>>(Qb, Kb, Vt, out);
    }
}
